// ASGLoss_15951508537382
// MI455X (gfx1250) — matrix operand order repaired and checked by reading, not compiled, not run on hardware
//
#include <hip/hip_runtime.h>
#include <hip/hip_bf16.h>
#include <math.h>

typedef __attribute__((ext_vector_type(16))) _Float16 v16h;
typedef __attribute__((ext_vector_type(8)))  _Float16 v8h;
typedef __attribute__((ext_vector_type(8)))  float    v8f;

#define ASG_B 64
#define ASG_T 1000
#define ASG_N 256
#define ASG_L 128
#define NEGINF (-1e30f)

#define LOG2E 1.4426950408889634f
#define LN2   0.6931471805599453f

#define MTILE 16
#define A_PAD 260   // alpha row stride in floats (bank-conflict padding)
#define P_PAD 264   // p row stride in halfs; 264*2 = 528 bytes, 16B aligned rows

// Raw hardware transcendentals (v_exp_f32 / v_log_f32 are base-2).
// Safe here: exp2 args are <= 0 (shifted by rowmax); log args in [0.9, 257].
__device__ __forceinline__ float fast_exp(float x) {
    return __builtin_amdgcn_exp2f(x * LOG2E);
}
__device__ __forceinline__ float fast_log(float x) {
    return LN2 * __builtin_amdgcn_logf(x);
}

// ---------------------------------------------------------------------------
// FCC: alpha[b,n] recurrence as f16 WMMA GEMM in the shifted-exp domain.
//   alpha'[b,n] = x[b,t,n] + m_b + log( sum_m exp(alpha[b,m]-m_b)*exp(trans[n,m]) )
// Grid: 4 blocks (16-batch M-tiles), 512 threads = 16 waves, wave w owns
// N-columns [16w, 16w+16). B operand exp(trans) for the wave's tile is held in
// registers (8 x v16h) across the whole t loop. Emission loads for step t are
// issued at the top of the iteration so their latency overlaps the rowmax
// reduction, the barrier, and the WMMA chain.
// ---------------------------------------------------------------------------
__global__ __launch_bounds__(512)
void asg_fcc_kernel(const float* __restrict__ x,
                    const float* __restrict__ trans,
                    float* __restrict__ fcc_out)
{
    __shared__ float    alpha[MTILE][A_PAD];
    __shared__ _Float16 p[MTILE][P_PAD];
    __shared__ float    maxv[MTILE];

    const int tid  = threadIdx.x;
    const int lane = tid & 31;
    const int wave = tid >> 5;      // 0..15 -> N tile id
    const int hi   = lane >> 4;     // lane half (0/1)
    const int lm   = lane & 15;
    const int b0   = blockIdx.x * MTILE;

    // --- Preload B fragments: B[k][n] = exp(trans[nGlobal, kGlobal]) ---
    // f16 B 32x16 operand, contiguous k order (this code's choice): lanes 0-15 col=lm hold K=0..15 (h==K),
    // lanes 16-31 col=lm hold K=16..31 (h==K-16).
    union BF { v16h v; _Float16 h[16]; };
    BF bfrag[8];
    {
        const int ncol = wave * 16 + lm;
        #pragma unroll
        for (int k0 = 0; k0 < 8; ++k0) {
            #pragma unroll
            for (int h = 0; h < 16; ++h) {
                const int m = k0 * 32 + hi * 16 + h;
                bfrag[k0].h[h] = (_Float16)__expf(trans[ncol * ASG_N + m]);
            }
        }
    }

    // Per-lane emission base: row r of the C tile is batch b0+r+8*hi, col ncol.
    const int ncol = wave * 16 + lm;
    const float* xbase = x + (size_t)(b0 + 8 * hi) * ASG_T * ASG_N + ncol;

    // --- alpha init: alpha = x[:, 0, :] for this M-tile ---
    for (int i = tid; i < MTILE * ASG_N; i += 512) {
        const int r = i >> 8, c = i & 255;
        alpha[r][c] = x[(size_t)(b0 + r) * ASG_T * ASG_N + c];
    }
    __syncthreads();

    for (int t = 1; t < ASG_T; ++t) {
        // --- Issue this step's 8 emission loads up front (distinct dests) ---
        float xv[8];
        #pragma unroll
        for (int r = 0; r < 8; ++r)
            xv[r] = xbase[(size_t)r * ASG_T * ASG_N + (size_t)t * ASG_N];

        // --- Phase 1: wave w reduces row w: rowmax + exp shift into p (f16) ---
        {
            const int w = wave;
            float vals[8];
            float mymax = NEGINF;
            #pragma unroll
            for (int i = 0; i < 8; ++i) {
                vals[i] = alpha[w][lane + 32 * i];
                mymax = fmaxf(mymax, vals[i]);
            }
            #pragma unroll
            for (int off = 16; off > 0; off >>= 1)
                mymax = fmaxf(mymax, __shfl_xor(mymax, off, 32));
            #pragma unroll
            for (int i = 0; i < 8; ++i)
                p[w][lane + 32 * i] = (_Float16)fast_exp(vals[i] - mymax);
            if (lane == 0) maxv[w] = mymax;
        }
        __syncthreads();

        // --- Phase 2: acc[b][n] = sum_m p[b][m] * exp(trans[n][m]) via WMMA ---
        v8f acc = {};
        #pragma unroll
        for (int k0 = 0; k0 < 8; ++k0) {
            // f16 A 16x32 operand, same contiguous k order as B: lane half hi, row lm;
            // halfs 0..7  <-> K = hi*16 + 0..7
            // halfs 8..15 <-> K = hi*16 + 8..15   (16B-aligned LDS chunks)
            union AF { v16h v; v8h q[2]; } af;
            af.q[0] = *(const v8h*)&p[lm][k0 * 32 + hi * 16];
            af.q[1] = *(const v8h*)&p[lm][k0 * 32 + hi * 16 + 8];
            acc = __builtin_amdgcn_wmma_f32_16x16x32_f16(
                false, af.v, false, bfrag[k0].v, (short)0, acc, false, false);
        }

        // --- writeback: C/D layout -> row = r + 8*hi, col = ncol ---
        #pragma unroll
        for (int r = 0; r < 8; ++r) {
            const int row = r + 8 * hi;
            const float nv = xv[r] + maxv[row] + fast_log(acc[r]);
            alpha[row][ncol] = nv;
            __builtin_prefetch(&xbase[(size_t)r * ASG_T * ASG_N
                                      + (size_t)(t + 1) * ASG_N], 0, 3);
        }
        __syncthreads();
    }

    // --- final: fcc[b] = logsumexp_n alpha[b, n]; wave w handles row w ---
    {
        const int w = wave;
        float vals[8];
        float mymax = NEGINF;
        #pragma unroll
        for (int i = 0; i < 8; ++i) {
            vals[i] = alpha[w][lane + 32 * i];
            mymax = fmaxf(mymax, vals[i]);
        }
        #pragma unroll
        for (int off = 16; off > 0; off >>= 1)
            mymax = fmaxf(mymax, __shfl_xor(mymax, off, 32));
        float s = 0.f;
        #pragma unroll
        for (int i = 0; i < 8; ++i)
            s += fast_exp(vals[i] - mymax);
        #pragma unroll
        for (int off = 16; off > 0; off >>= 1)
            s += __shfl_xor(s, off, 32);
        if (lane == 0) fcc_out[b0 + w] = mymax + fast_log(s);
    }
}

// ---------------------------------------------------------------------------
// FAC: forced alignment over the target sequence. One block per utterance,
// thread j owns alignment position j.
// ---------------------------------------------------------------------------
__global__ __launch_bounds__(ASG_L)
void asg_fac_kernel(const float* __restrict__ x,
                    const int*   __restrict__ target,
                    const int*   __restrict__ target_size,
                    const float* __restrict__ trans,
                    float* __restrict__ fac_out)
{
    __shared__ float al[ASG_L];
    const int b = blockIdx.x;
    const int j = threadIdx.x;

    const int tj = target[b * ASG_L + j];
    const float tself = trans[tj * ASG_N + tj];
    const float tmove = (j > 0)
        ? trans[tj * ASG_N + target[b * ASG_L + j - 1]]
        : NEGINF;

    const float* xb = x + (size_t)b * ASG_T * ASG_N;

    al[j] = (j == 0) ? xb[tj] : NEGINF;
    __syncthreads();

    float nv = al[j];
    for (int t = 1; t < ASG_T; ++t) {
        const float e = xb[(size_t)t * ASG_N + tj];   // issue load first
        __builtin_prefetch(&xb[(size_t)(t + 1) * ASG_N + tj], 0, 3);
        const float prev = al[j];
        const float pm1  = (j > 0) ? al[j - 1] : NEGINF;
        const float stay = prev + tself;
        const float mv   = pm1 + tmove;
        const float m    = fmaxf(stay, mv);
        const float d    = fminf(stay, mv) - m;       // <= 0
        // logaddexp via raw exp2/log2: m + ln2*log2(1 + 2^(d*log2e))
        nv = m + fast_log(1.0f + fast_exp(d)) + e;
        __syncthreads();
        al[j] = nv;
        __syncthreads();
    }

    if (j == target_size[b] - 1) fac_out[b] = nv;
}

// ---------------------------------------------------------------------------
__global__ void asg_combine_kernel(const float* __restrict__ fcc,
                                   const float* __restrict__ fac,
                                   float* __restrict__ out)
{
    const int b = threadIdx.x;
    if (b < ASG_B) out[b] = fcc[b] - fac[b];
}

extern "C" void kernel_launch(void* const* d_in, const int* in_sizes, int n_in,
                              void* d_out, int out_size, void* d_ws, size_t ws_size,
                              hipStream_t stream) {
    (void)in_sizes; (void)n_in; (void)out_size; (void)ws_size;
    const float* x           = (const float*)d_in[0];   // [B, T, N] f32
    const int*   target      = (const int*)  d_in[1];   // [B, L] i32
    const int*   target_size = (const int*)  d_in[2];   // [B] i32
    const float* trans       = (const float*)d_in[3];   // [N, N] f32
    float* out = (float*)d_out;                         // [B] f32

    float* fcc = (float*)d_ws;          // 64 floats
    float* fac = fcc + ASG_B;           // 64 floats

    asg_fcc_kernel<<<ASG_B / MTILE, 512, 0, stream>>>(x, trans, fcc);
    asg_fac_kernel<<<ASG_B, ASG_L, 0, stream>>>(x, target, target_size, trans, fac);
    asg_combine_kernel<<<1, ASG_B, 0, stream>>>(fcc, fac, out);
}
